// AnyPrecisionLinear_9028021256799
// MI455X (gfx1250) — compile-verified, matrix operand order checked by reading, not run on hardware
//
#include <hip/hip_runtime.h>

typedef __attribute__((ext_vector_type(16))) _Float16 v16h;
typedef __attribute__((ext_vector_type(8)))  _Float16 v8h;
typedef __attribute__((ext_vector_type(8)))  float    v8f;
typedef int v4i __attribute__((vector_size(16)));

#define K_DIM 4096
#define O_DIM 4096
#define GWORDS (K_DIM / 32)            // 128 int32 words per output row per plane
#define PLANE_STRIDE (O_DIM * GWORDS)  // 524288 words per bitplane

#define BM 256
#define BN 128
#define BK 32
#define KTILES (K_DIM / BK)            // 128

// fp16 max * (1 - 0.005)
#define CLAMP_V 65176.48f

#if __has_builtin(__builtin_amdgcn_global_load_async_to_lds_b128)
#define HAVE_ASYNC_LDS 1
#else
#define HAVE_ASYNC_LDS 0
#endif

// 16-byte global -> LDS copy: async DMA path on CDNA5 if available.
__device__ __forceinline__ void copy16(const _Float16* g, _Float16* l) {
#if HAVE_ASYNC_LDS
    __builtin_amdgcn_global_load_async_to_lds_b128(
        (__attribute__((address_space(1))) v4i*)g,
        (__attribute__((address_space(3))) v4i*)l, 0, 0);
#else
    *(v8h*)l = *(const v8h*)g;
#endif
}

__device__ __forceinline__ void wait_async_all() {
#if HAVE_ASYNC_LDS
#if __has_builtin(__builtin_amdgcn_s_wait_asynccnt)
    __builtin_amdgcn_s_wait_asynccnt(0);
#else
    asm volatile("s_wait_asynccnt 0x0" ::: "memory");
#endif
#endif
}

// ---------------------------------------------------------------------------
// Kernel 1: bitplane-LUT dequant.  qweight (4, O, K/32) int32, plane 0 = MSB.
// Each block handles 2 output rows (2 LUT rows cached in LDS); each thread
// expands one 32-bit word group into 32 fp16 weights and stores 4x b128.
// ---------------------------------------------------------------------------
__global__ __launch_bounds__(256) void dequant_kernel(
    const int* __restrict__ qw, const _Float16* __restrict__ lut,
    _Float16* __restrict__ W)
{
    __shared__ __align__(16) _Float16 slut[2 * 16];
    const int tid   = threadIdx.x;
    const int obase = blockIdx.x * 2;

    if (tid < 32) slut[tid] = lut[obase * 16 + tid];
    __syncthreads();

    const int orow = tid >> 7;            // 0 or 1 (128 threads per output row)
    const int o    = obase + orow;
    const int g    = tid & 127;           // word index within row
    const int idx  = o * GWORDS + g;

    const unsigned p0 = (unsigned)qw[0 * PLANE_STRIDE + idx]; // MSB plane
    const unsigned p1 = (unsigned)qw[1 * PLANE_STRIDE + idx];
    const unsigned p2 = (unsigned)qw[2 * PLANE_STRIDE + idx];
    const unsigned p3 = (unsigned)qw[3 * PLANE_STRIDE + idx];

    const _Float16* lrow = &slut[orow * 16];
    _Float16* dst = W + (size_t)o * K_DIM + g * 32;

#pragma unroll
    for (int v = 0; v < 4; ++v) {
        v8h pack;
#pragma unroll
        for (int e = 0; e < 8; ++e) {
            const int b = v * 8 + e;
            const int code = (int)((((p0 >> b) & 1u) << 3) |
                                   (((p1 >> b) & 1u) << 2) |
                                   (((p2 >> b) & 1u) << 1) |
                                   ((p3 >> b) & 1u));
            pack[e] = lrow[code];          // ds_load_u16 from LDS LUT
        }
        *(v8h*)(dst + v * 8) = pack;       // global_store_b128
    }
}

// ---------------------------------------------------------------------------
// Kernel 2: fp16 WMMA GEMM  Y[M,O] = X[M,K] * W[O,K]^T + bias, clamped.
// Block tile 256x128, BK=32, double-buffered LDS (48 KB), 8 waves (4 M x 2 N),
// wave tile 64x64 = 4x4 fragments; all 8 fragments loaded up-front so the
// 16 v_wmma_f32_16x16x32_f16 per K-step can issue back-to-back.
// ---------------------------------------------------------------------------
__global__ __launch_bounds__(256) void gemm_kernel(
    const _Float16* __restrict__ X,     // M x K row-major
    const _Float16* __restrict__ W,     // O x K row-major (dequantized)
    const _Float16* __restrict__ bias,  // O
    float* __restrict__ Y)              // M x O row-major (f32 out)
{
    __shared__ __align__(16) _Float16 sA[2][BM][BK];
    __shared__ __align__(16) _Float16 sB[2][BN][BK];

    const int tid  = threadIdx.x;
    const int lane = tid & 31;
    const int wave = tid >> 5;
    const int wm   = wave & 3;           // 0..3  -> M offset wm*64
    const int wn   = wave >> 2;          // 0..1  -> N offset wn*64

    const int m0 = blockIdx.y * BM;
    const int n0 = blockIdx.x * BN;

    const _Float16* Abase = X + (size_t)m0 * K_DIM;
    const _Float16* Bbase = W + (size_t)n0 * K_DIM;

    // fragment addressing per ISA 16-bit operand layout
    const int half = lane >> 4;          // 0: K 0-7 / 16-23,  1: K 8-15 / 24-31
    const int fr   = lane & 15;          // row (A) / col (B) within fragment
    const int kfo  = half * 8;

    v8f acc[4][4] = {};

    // ---- staging: A tile = 1024 16B chunks (4/thread), B tile = 512 (2/thread)
    auto stage_tile = [&](int kt, int buf) {
        const _Float16* Ak = Abase + (size_t)kt * BK;
        const _Float16* Bk = Bbase + (size_t)kt * BK;
#pragma unroll
        for (int i = 0; i < 4; ++i) {
            const int c = tid + 256 * i;
            const int r = c >> 2, s = (c & 3) * 8;
            copy16(Ak + (size_t)r * K_DIM + s, &sA[buf][r][s]);
        }
#pragma unroll
        for (int i = 0; i < 2; ++i) {
            const int c = tid + 256 * i;
            const int r = c >> 2, s = (c & 3) * 8;
            copy16(Bk + (size_t)r * K_DIM + s, &sB[buf][r][s]);
        }
    };

    stage_tile(0, 0);
    wait_async_all();
    __syncthreads();

    for (int kt = 0; kt < KTILES; ++kt) {
        const int cur = kt & 1;
        const int nxt = cur ^ 1;

        // kick off next tile's global->LDS transfers (other buffer)
        if (kt + 1 < KTILES) stage_tile(kt + 1, nxt);

        // ---- load ALL fragments first, then chain the 16 WMMAs ----
        v16h afr[4], bfr[4];
#pragma unroll
        for (int i = 0; i < 4; ++i) {
            const _Float16* ap = &sA[cur][wm * 64 + i * 16 + fr][kfo];
            v8h lo = *(const v8h*)ap;           // ds_load_b128
            v8h hi = *(const v8h*)(ap + 16);    // ds_load_b128
            afr[i] = __builtin_shufflevector(lo, hi,
                0, 1, 2, 3, 4, 5, 6, 7, 8, 9, 10, 11, 12, 13, 14, 15);
        }
#pragma unroll
        for (int j = 0; j < 4; ++j) {
            const _Float16* bp = &sB[cur][wn * 64 + j * 16 + fr][kfo];
            v8h lo = *(const v8h*)bp;
            v8h hi = *(const v8h*)(bp + 16);
            bfr[j] = __builtin_shufflevector(lo, hi,
                0, 1, 2, 3, 4, 5, 6, 7, 8, 9, 10, 11, 12, 13, 14, 15);
        }
#pragma unroll
        for (int i = 0; i < 4; ++i) {
#pragma unroll
            for (int j = 0; j < 4; ++j) {
                acc[i][j] = __builtin_amdgcn_wmma_f32_16x16x32_f16(
                    false, afr[i], false, bfr[j],
                    (short)0, acc[i][j], false, false);
            }
        }

        wait_async_all();     // this wave's async LDS writes are done
        __syncthreads();      // everyone's are visible
    }

    // ---- epilogue: bias + clamp + store f32 ----
    // C/D layout: VGPR r -> M = r + 8*(lane>=16), N = lane & 15
#pragma unroll
    for (int j = 0; j < 4; ++j) {
        const int n   = n0 + wn * 64 + j * 16 + fr;
        const float bv = (float)bias[n];
#pragma unroll
        for (int i = 0; i < 4; ++i) {
            const int mbase = m0 + wm * 64 + i * 16 + half * 8;
#pragma unroll
            for (int r = 0; r < 8; ++r) {
                float v = acc[i][j][r] + bv;
                v = fminf(fmaxf(v, -CLAMP_V), CLAMP_V);
                Y[(size_t)(mbase + r) * O_DIM + n] = v;
            }
        }
    }
}

extern "C" void kernel_launch(void* const* d_in, const int* in_sizes, int n_in,
                              void* d_out, int out_size, void* d_ws, size_t ws_size,
                              hipStream_t stream) {
    const _Float16* x    = (const _Float16*)d_in[0];  // (8,2048,4096) fp16
    const int*      qw   = (const int*)d_in[1];       // (4,4096,128) int32
    const _Float16* lut  = (const _Float16*)d_in[2];  // (4096,16) fp16
    const _Float16* bias = (const _Float16*)d_in[3];  // (4096,) fp16
    float*          out  = (float*)d_out;             // (8,2048,4096) f32

    _Float16* W = (_Float16*)d_ws;                    // 4096*4096*2 = 32 MB scratch

    const int M = in_sizes[0] / K_DIM;                // 16384

    // 1) dequant W once into workspace
    dequant_kernel<<<O_DIM / 2, 256, 0, stream>>>(qw, lut, W);

    // 2) WMMA GEMM
    dim3 grid(O_DIM / BN, M / BM);
    gemm_kernel<<<grid, 256, 0, stream>>>(x, W, bias, out);
}
